// SS2D_10264971837499
// MI455X (gfx1250) — compile-verified, matrix operand order checked by reading, not run on hardware
//
#include <hip/hip_runtime.h>
#include <hip/hip_bf16.h>

// ---------------------------------------------------------------------------
// SS2D (VMamba selective scan 2D) for gfx1250.
// B=4, H=W=128, L=16384, D_MODEL=128, D_INNER=256, DT_RANK=8, N(state)=1, K=4
// ---------------------------------------------------------------------------

typedef __attribute__((ext_vector_type(16))) _Float16 v16h;
typedef __attribute__((ext_vector_type(8)))  _Float16 v8h;
typedef __attribute__((ext_vector_type(8)))  float    v8f;

#define BATCH   4
#define HDIM    128
#define WDIM    128
#define LSEQ    16384      // H*W
#define DMODEL  128
#define DINNER  256
#define DTRANK  8
#define KDIR    4
#define NPROJ   10         // DT_RANK + 2*D_STATE

// Build a v16h A/B operand from two contiguous 16-byte chunks.
// ISA 16-bit operand layout: element pairs [2j,2j+1] cover K = kBase+8*half+{0..7}
// (j<4) and K = kBase+16+8*half+{0..7} (j>=4) -> exactly two b128 loads.
__device__ __forceinline__ v16h load_op16(const _Float16* __restrict__ p) {
    const v8h lo = *(const v8h*)(p);        // K offsets 0..7
    const v8h hi = *(const v8h*)(p + 16);   // K offsets 16..23
    return __builtin_shufflevector(lo, hi, 0, 1, 2, 3, 4, 5, 6, 7,
                                           8, 9, 10, 11, 12, 13, 14, 15);
}

// ---------------------------------------------------------------------------
// Precision-conversion prep kernels (operands become small & L2-resident).
// ---------------------------------------------------------------------------
__global__ void k_cvt_x(const float* __restrict__ x, _Float16* __restrict__ xh) {
    const size_t g = (size_t)blockIdx.x * 256 + threadIdx.x;   // B*L*DMODEL
    xh[g] = (_Float16)x[g];
}
// Win (128,512) -> Wt (512,128) f16  (row n = output column, K contiguous)
__global__ void k_cvt_wt(const float* __restrict__ Win, _Float16* __restrict__ Wt) {
    const int g = blockIdx.x * 256 + threadIdx.x;              // 512*128
    const int k = g & 127, n = g >> 7;
    Wt[n * DMODEL + k] = (_Float16)Win[(size_t)k * 512 + n];
}
// Wout (256,128) -> Wot (128,256) f16
__global__ void k_cvt_wot(const float* __restrict__ Wout, _Float16* __restrict__ Wot) {
    const int g = blockIdx.x * 256 + threadIdx.x;              // 128*256
    const int k = g & 255, n = g >> 8;
    Wot[n * DINNER + k] = (_Float16)Wout[(size_t)k * DMODEL + n];
}

// ---------------------------------------------------------------------------
// Kernel 1: xz = x @ Win.  xh:(B*L,128) f16, Wt:(512,128) f16 (pre-transposed).
// One wave per 16x16 C tile; 4x v_wmma_f32_16x16x32_f16 along K=128.
// Destination (xb vs z) is wave-uniform -> single pointer select, no exec math.
// ---------------------------------------------------------------------------
__global__ void k_in_gemm(const _Float16* __restrict__ A, const _Float16* __restrict__ Bt,
                          float* __restrict__ xb, float* __restrict__ z) {
    const int lane = threadIdx.x & 31;
    const int wave = threadIdx.x >> 5;
    const int tile = blockIdx.x * 8 + wave;         // 4096*32 tiles
    const int mT = tile >> 5;
    const int nT = tile & 31;
    const int mBase = mT * 16, nBase = nT * 16;
    const int half = lane >> 4;
    const int ml   = lane & 15;

    const _Float16* __restrict__ Arow = A  + (size_t)(mBase + ml) * DMODEL + 8 * half;
    const _Float16* __restrict__ Brow = Bt + (size_t)(nBase + ml) * DMODEL + 8 * half;

    v8f acc = {};
    #pragma unroll
    for (int kk = 0; kk < 4; ++kk) {
        const v16h av = load_op16(Arow + kk * 32);
        const v16h bv = load_op16(Brow + kk * 32);
        acc = __builtin_amdgcn_wmma_f32_16x16x32_f16(false, av, false, bv,
                                                     (short)0, acc, false, false);
    }
    // wave-uniform destination select (nBase is uniform, multiple of 16)
    float* __restrict__ dst = (nBase < DINNER) ? (xb + nBase) : (z + (nBase - DINNER));
    #pragma unroll
    for (int v = 0; v < 8; ++v) {
        const int row = mBase + v + 8 * half;
        dst[(size_t)row * DINNER + ml] = acc[v];
    }
}

// ---------------------------------------------------------------------------
// Kernel 2: depthwise 3x3 conv (pad 1) + bias + SiLU.
// Input xb (B,H,W,C); output xconv (B,L,C) — channel-fastest for coalescing.
// ---------------------------------------------------------------------------
__global__ void k_conv(const float* __restrict__ xb, const float* __restrict__ cw,
                       const float* __restrict__ cb, float* __restrict__ xconv) {
    const size_t g = (size_t)blockIdx.x * 256 + threadIdx.x;   // B*L*C
    const int c = (int)(g & 255);
    size_t s = g >> 8;
    const int w = (int)(s & 127); s >>= 7;
    const int h = (int)(s & 127); s >>= 7;
    const int b = (int)s;
    float acc = 0.f;
    #pragma unroll
    for (int di = -1; di <= 1; ++di) {
        const int hh = h + di;
        if (hh < 0 || hh >= HDIM) continue;
        #pragma unroll
        for (int dj = -1; dj <= 1; ++dj) {
            const int ww = w + dj;
            if (ww < 0 || ww >= WDIM) continue;
            acc += xb[(((size_t)(b * HDIM + hh) * WDIM) + ww) * DINNER + c]
                 * cw[c * 9 + (di + 1) * 3 + (dj + 1)];
        }
    }
    acc += cb[c];
    xconv[g] = acc / (1.f + expf(-acc));   // SiLU
}

// ---------------------------------------------------------------------------
// Kernel 3: zero y_acc (d_ws is poisoned to 0xAA by the harness).
// ---------------------------------------------------------------------------
__global__ void k_zero(float* __restrict__ y) {
    y[(size_t)blockIdx.x * 256 + threadIdx.x] = 0.f;
}

// ---------------------------------------------------------------------------
// Kernel 4: direction-independent projection  proj(B,L,10) = xconv @ Wxproj.
// One wave per (b,l) position; lanes stride channels, wave-reduce via shfl.
// ---------------------------------------------------------------------------
__global__ void k_proj(const float* __restrict__ xconv, const float* __restrict__ Wx,
                       float* __restrict__ proj) {
    const int lane = threadIdx.x & 31;
    const int wave = threadIdx.x >> 5;
    const int p = blockIdx.x * 8 + wave;            // 0..B*L-1
    const float* __restrict__ xr = xconv + (size_t)p * DINNER;
    float acc[NPROJ];
    #pragma unroll
    for (int d = 0; d < NPROJ; ++d) acc[d] = 0.f;
    for (int c = lane; c < DINNER; c += 32) {
        const float xv = xr[c];
        #pragma unroll
        for (int d = 0; d < NPROJ; ++d) acc[d] += xv * Wx[c * NPROJ + d];
    }
    #pragma unroll
    for (int d = 0; d < NPROJ; ++d)
        for (int m = 16; m >= 1; m >>= 1)
            acc[d] += __shfl_xor(acc[d], m, 32);
    if (lane == 0) {
        float* __restrict__ pr = proj + (size_t)p * NPROJ;
        #pragma unroll
        for (int d = 0; d < NPROJ; ++d) pr[d] = acc[d];
    }
}

// ---------------------------------------------------------------------------
// Kernel 5: selective scan, D_STATE=1 -> scalar recurrence per (k,b,c).
// ---------------------------------------------------------------------------
__device__ __forceinline__ int dir_pos(int k, int l) {
    if (k == 0) return l;
    if (k == 1) return LSEQ - 1 - l;
    if (k == 2) return ((l & 127) << 7) | (l >> 7);
    const int lr = LSEQ - 1 - l;
    return ((lr & 127) << 7) | (lr >> 7);
}

__global__ void k_scan(const float* __restrict__ xconv, const float* __restrict__ proj,
                       const float* __restrict__ dtw_g, const float* __restrict__ A_log,
                       const float* __restrict__ D_param, const float* __restrict__ dt_bias,
                       float* __restrict__ y_acc) {
    const int g = blockIdx.x * 256 + threadIdx.x;   // K*B*C = 16384
    const int c = g & 255;
    const int b = (g >> 8) & 3;
    const int k = g >> 10;
    const int kc = k * DINNER + c;

    float dtw[DTRANK];
    #pragma unroll
    for (int r = 0; r < DTRANK; ++r) dtw[r] = dtw_g[kc * DTRANK + r];
    const float Av = -expf(A_log[kc]);              // D_STATE == 1
    const float Dv = D_param[kc];
    const float db = dt_bias[kc];
    const size_t bbase = (size_t)b * LSEQ;

    float state = 0.f;
    for (int l = 0; l < LSEQ; ++l) {
        const size_t base = bbase + dir_pos(k, l);
        // software prefetch for the permuted walk (global_prefetch_b8)
        if (l + 4 < LSEQ) {
            const size_t nb = bbase + dir_pos(k, l + 4);
            __builtin_prefetch(xconv + nb * DINNER + c, 0, 1);
            __builtin_prefetch(proj + nb * NPROJ, 0, 1);
        }
        const float u = xconv[base * DINNER + c];
        const float* __restrict__ pr = proj + base * NPROJ;  // broadcast across wave
        float draw = db;
        #pragma unroll
        for (int r = 0; r < DTRANK; ++r) draw += dtw[r] * pr[r];
        const float delta = (draw > 20.f) ? draw : log1pf(expf(draw));
        const float dA = expf(delta * Av);
        state = dA * state + delta * u * pr[8];
        const float y = state * pr[9] + u * Dv;
        atomicAdd(&y_acc[base * DINNER + c], y);
    }
}

// ---------------------------------------------------------------------------
// Kernel 6: gate with SiLU(z), LayerNorm over channels, store f16 for out-GEMM.
// ---------------------------------------------------------------------------
__global__ void k_gate_ln(const float* __restrict__ y_acc, const float* __restrict__ z,
                          const float* __restrict__ gamma, const float* __restrict__ beta,
                          _Float16* __restrict__ yn) {
    __shared__ float red[256];
    const int c = threadIdx.x;
    const size_t idx = (size_t)blockIdx.x * DINNER + c;
    const float zv = z[idx];
    const float yz = y_acc[idx] * (zv / (1.f + expf(-zv)));

    red[c] = yz;
    __syncthreads();
    for (int s = 128; s > 0; s >>= 1) {
        if (c < s) red[c] += red[c + s];
        __syncthreads();
    }
    const float mu = red[0] * (1.f / DINNER);
    __syncthreads();
    const float xc = yz - mu;
    red[c] = xc * xc;
    __syncthreads();
    for (int s = 128; s > 0; s >>= 1) {
        if (c < s) red[c] += red[c + s];
        __syncthreads();
    }
    const float var = red[0] * (1.f / DINNER);
    const float out = gamma[c] * xc * rsqrtf(var + 1e-5f) + beta[c];
    yn[idx] = (_Float16)out;
}

// ---------------------------------------------------------------------------
// Kernel 7: out = yn @ Wout.  yn:(B*L,256) f16, Wot:(128,256) f16 pre-transposed.
// One wave per 16x16 tile, 8x v_wmma_f32_16x16x32_f16 along K=256.
// ---------------------------------------------------------------------------
__global__ void k_out_gemm(const _Float16* __restrict__ A, const _Float16* __restrict__ Bt,
                           float* __restrict__ out) {
    const int lane = threadIdx.x & 31;
    const int wave = threadIdx.x >> 5;
    const int tile = blockIdx.x * 8 + wave;         // 4096*8 tiles
    const int mT = tile >> 3;
    const int nT = tile & 7;
    const int mBase = mT * 16, nBase = nT * 16;
    const int half = lane >> 4;
    const int ml   = lane & 15;

    const _Float16* __restrict__ Arow = A  + (size_t)(mBase + ml) * DINNER + 8 * half;
    const _Float16* __restrict__ Brow = Bt + (size_t)(nBase + ml) * DINNER + 8 * half;

    v8f acc = {};
    #pragma unroll
    for (int kk = 0; kk < 8; ++kk) {
        const v16h av = load_op16(Arow + kk * 32);
        const v16h bv = load_op16(Brow + kk * 32);
        acc = __builtin_amdgcn_wmma_f32_16x16x32_f16(false, av, false, bv,
                                                     (short)0, acc, false, false);
    }
    #pragma unroll
    for (int v = 0; v < 8; ++v) {
        const int row = mBase + v + 8 * half;
        out[(size_t)row * DMODEL + nBase + ml] = acc[v];
    }
}

// ---------------------------------------------------------------------------
// Launcher
// ---------------------------------------------------------------------------
extern "C" void kernel_launch(void* const* d_in, const int* in_sizes, int n_in,
                              void* d_out, int out_size, void* d_ws, size_t ws_size,
                              hipStream_t stream) {
    const float* x        = (const float*)d_in[0];
    const float* Win      = (const float*)d_in[1];
    const float* conv_w   = (const float*)d_in[2];
    const float* conv_b   = (const float*)d_in[3];
    const float* Wxproj   = (const float*)d_in[4];
    const float* dt_w     = (const float*)d_in[5];
    const float* A_log    = (const float*)d_in[6];
    const float* D_param  = (const float*)d_in[7];
    const float* dt_bias  = (const float*)d_in[8];
    const float* gamma    = (const float*)d_in[9];
    const float* beta     = (const float*)d_in[10];
    const float* Wout     = (const float*)d_in[11];
    float* out = (float*)d_out;

    // Workspace layout (bytes). y_acc aliases xb (dead after the conv).
    char* ws = (char*)d_ws;
    const size_t BLC = (size_t)BATCH * LSEQ * DINNER;          // 16.7M elems
    float*    xb    = (float*)(ws);                            // 67.1 MB
    float*    y_acc = xb;                                      // alias (reuse)
    float*    zbuf  = (float*)(ws + BLC * 4);                  // 67.1 MB
    float*    xconv = (float*)(ws + BLC * 8);                  // 67.1 MB
    float*    proj  = (float*)(ws + BLC * 12);                 //  2.6 MB
    _Float16* yn    = (_Float16*)(ws + BLC * 12 + (size_t)BATCH * LSEQ * NPROJ * 4);
    char*     ws2   = ws + BLC * 12 + (size_t)BATCH * LSEQ * NPROJ * 4 + BLC * 2;
    _Float16* xh    = (_Float16*)(ws2);                        // 16.8 MB
    _Float16* Wt    = (_Float16*)(ws2 + (size_t)BATCH * LSEQ * DMODEL * 2);
    _Float16* Wot   = (_Float16*)(ws2 + (size_t)BATCH * LSEQ * DMODEL * 2 + 512 * 128 * 2);

    // 0. f16 operand prep (tiny, L2-resident)
    k_cvt_x  <<<32768, 256, 0, stream>>>(x, xh);               // 8.39M elems
    k_cvt_wt <<<256,   256, 0, stream>>>(Win, Wt);
    k_cvt_wot<<<128,   256, 0, stream>>>(Wout, Wot);
    // 1. input GEMM (WMMA):   (65536x128)@(128x512) -> xb | z
    k_in_gemm<<<16384, 256, 0, stream>>>(xh, Wt, xb, zbuf);
    // 2. depthwise 3x3 conv + SiLU, (B,H,W,C) -> (B,L,C)
    k_conv<<<65536, 256, 0, stream>>>(xb, conv_w, conv_b, xconv);
    // 3. zero the accumulation buffer (aliases xb, now dead)
    k_zero<<<65536, 256, 0, stream>>>(y_acc);
    // 4. shared projection (dt | B | C), computed once for all 4 directions
    k_proj<<<8192, 256, 0, stream>>>(xconv, Wxproj, proj);
    // 5. 4-direction selective scan with inverse-permutation atomic accumulate
    k_scan<<<64, 256, 0, stream>>>(xconv, proj, dt_w, A_log, D_param, dt_bias, y_acc);
    // 6. SiLU gate + LayerNorm, emit f16 activations
    k_gate_ln<<<65536, 256, 0, stream>>>(y_acc, zbuf, gamma, beta, yn);
    // 7. output GEMM (WMMA): (65536x256)@(256x128) -> out
    k_out_gemm<<<4096, 256, 0, stream>>>(yn, Wot, out);
}
